// TriAttention_34857954574979
// MI455X (gfx1250) — compile-verified, matrix operand order checked by reading, not run on hardware
//
#include <hip/hip_runtime.h>

// ---------------------------------------------------------------------------
// TriAttention for MI455X (gfx1250): bf16 WMMA, flash-style (no SxS in HBM).
//   proj:   v/t/a = x @ W + b            (bf16 WMMA, f32 accum)
//   stats:  per-row (max, sumexp) of the 3 logit matrices (vt, va, ta)
//   fused:  f_x = resid + a1*attn1@X1 + a2*attn2@X2, weights finalized from
//           precomputed stats (indexed by out-row for forward products, by
//           streamed column for transposed products)
// Round-2 changes: Q fragments register-cached across the whole stream
// (invariant), LDS transpose rewritten as batched b32 loads + b128 stores
// (single dscnt wait, no v_mov_b16 chains), next-tile prefetch.
// Workspace use: ~76 MB (3x fp32 proj, 3x bf16 proj, 6x [B*S] stats).
// ---------------------------------------------------------------------------

typedef __bf16 bf16;
typedef __attribute__((ext_vector_type(16))) __bf16 bf16x16;
typedef __attribute__((ext_vector_type(8)))  __bf16 bf16x8;
typedef __attribute__((ext_vector_type(4)))  __bf16 bf16x4;
typedef __attribute__((ext_vector_type(8)))  float  f32x8;

#define S_LEN 2048
#define DDIM  256
#define QSTR  264   // padded row stride (halves) for [rows][256] LDS tiles
#define TSTR  40    // padded row stride (halves) for [rows][32] LDS tiles

// Load a 16x32 bf16 fragment (A layout, or B^T stored row-major) from LDS.
// Lane l (l&15 = row, l>>4 = K-group): halves 0..7 = K=(8*kg..8*kg+7),
// halves 8..15 = K=(16+8*kg..23+8*kg)  — matches ISA 16-bit A/B layout.
__device__ __forceinline__ bf16x16 load_frag(const bf16* base, int row0, int k0,
                                             int stride) {
  const int lane = threadIdx.x & 31;
  const int lm = lane & 15, kg = lane >> 4;
  const bf16* p = base + (row0 + lm) * stride + k0 + kg * 8;
  bf16x8 lo = *(const bf16x8*)p;
  bf16x8 hi = *(const bf16x8*)(p + 16);
  bf16x16 r;
#pragma unroll
  for (int i = 0; i < 8; ++i) { r[i] = lo[i]; r[i + 8] = hi[i]; }
  return r;
}

__device__ __forceinline__ f32x8 wmma_bf16(bf16x16 a, bf16x16 b, f32x8 c) {
  return __builtin_amdgcn_wmma_f32_16x16x32_bf16(false, a, false, b,
                                                 (short)0, c, false, false);
}

// Transpose a [32 x 256] bf16 row-major LDS tile (stride QSTR) into a
// [256 x 32] tile (stride TSTR). Each thread handles a column-pair x 8-row
// sliver: 8 independent b32 loads (one wait), integer repack, 2 b128 stores.
__device__ __forceinline__ void transpose_tile(const bf16* src, bf16* dst) {
  const int tid = threadIdx.x;
#pragma unroll
  for (int it = 0; it < 2; ++it) {
    int v = it * 256 + tid;       // 0..511
    int cp = v & 127;             // column-pair index
    int jg = v >> 7;              // 8-row group (0..3)
    int c = cp * 2;
    unsigned t[8];
#pragma unroll
    for (int e = 0; e < 8; ++e)
      t[e] = *(const unsigned*)&src[(jg * 8 + e) * QSTR + c];
    uint4 lo, hi;
    lo.x = (t[0] & 0xffffu) | (t[1] << 16);
    lo.y = (t[2] & 0xffffu) | (t[3] << 16);
    lo.z = (t[4] & 0xffffu) | (t[5] << 16);
    lo.w = (t[6] & 0xffffu) | (t[7] << 16);
    hi.x = (t[0] >> 16) | (t[1] & 0xffff0000u);
    hi.y = (t[2] >> 16) | (t[3] & 0xffff0000u);
    hi.z = (t[4] >> 16) | (t[5] & 0xffff0000u);
    hi.w = (t[6] >> 16) | (t[7] & 0xffff0000u);
    *(uint4*)&dst[(c + 0) * TSTR + jg * 8] = lo;
    *(uint4*)&dst[(c + 1) * TSTR + jg * 8] = hi;
  }
}

// ---------------------------------------------------------------------------
// Projection: P = In[BS,Din] @ W[Din,256] + bias. Writes fp32 + bf16 copies.
// Grid: BS/64 blocks of 256 threads (8 waves: 4 row-blocks x 2 col-halves).
// ---------------------------------------------------------------------------
__global__ __launch_bounds__(256) void proj_kernel(
    const float* __restrict__ In, const float* __restrict__ W,
    const float* __restrict__ bias, int Din,
    float* __restrict__ P32, bf16* __restrict__ Pbf) {
  __shared__ bf16 sA[64 * TSTR];      // 64 x 32 A tile (bf16)
  __shared__ bf16 sW[32 * QSTR];      // 32 x 256 W rows (staging)
  __shared__ bf16 sWT[DDIM * TSTR];   // 256 x 32 W^T (B fragments)

  const int tid = threadIdx.x;
  const int wave = tid >> 5, lane = tid & 31;
  const int lm = lane & 15, kg = lane >> 4;
  const int wm = wave >> 1, wn = wave & 1;
  const int m0 = blockIdx.x * 64;

  f32x8 acc[8] = {};

  for (int k0 = 0; k0 < Din; k0 += 32) {
    __syncthreads();
    // A tile: 64 rows x 32 cols, fp32 -> bf16
    {
      int r = tid >> 2, c = (tid & 3) * 8;
      const float* p = In + (size_t)(m0 + r) * Din + k0 + c;
      float4 f0 = *(const float4*)p;
      float4 f1 = *(const float4*)(p + 4);
      bf16x8 h;
      h[0] = (bf16)f0.x; h[1] = (bf16)f0.y; h[2] = (bf16)f0.z; h[3] = (bf16)f0.w;
      h[4] = (bf16)f1.x; h[5] = (bf16)f1.y; h[6] = (bf16)f1.z; h[7] = (bf16)f1.w;
      *(bf16x8*)&sA[r * TSTR + c] = h;
    }
    // W rows k0..k0+31 (coalesced), fp32 -> bf16, row-major staging
#pragma unroll
    for (int it = 0; it < 8; ++it) {
      int flat = it * 1024 + tid * 4;
      int r = flat >> 8, c = flat & 255;
      float4 f = *(const float4*)(W + (size_t)(k0 + r) * DDIM + c);
      bf16x4 h;
      h[0] = (bf16)f.x; h[1] = (bf16)f.y; h[2] = (bf16)f.z; h[3] = (bf16)f.w;
      *(bf16x4*)&sW[r * QSTR + c] = h;
    }
    // prefetch next W tile while this one is consumed
    if (k0 + 32 < Din)
      __builtin_prefetch(W + (size_t)(k0 + 32) * DDIM + tid * 32, 0, 1);
    __syncthreads();
    transpose_tile(sW, sWT);
    __syncthreads();
    bf16x16 afr = load_frag(sA, 16 * wm, 0, TSTR);
#pragma unroll
    for (int t = 0; t < 8; ++t) {
      bf16x16 bfr = load_frag(sWT, 128 * wn + 16 * t, 0, TSTR);
      acc[t] = wmma_bf16(afr, bfr, acc[t]);
    }
  }
  // epilogue: bias + dual-precision writeback (C layout: VGPR r -> M, lane -> N)
#pragma unroll
  for (int t = 0; t < 8; ++t) {
#pragma unroll
    for (int r = 0; r < 8; ++r) {
      int row = m0 + 16 * wm + r + 8 * kg;
      int col = 128 * wn + 16 * t + lm;
      float v = acc[t][r] + bias[col];
      P32[(size_t)row * DDIM + col] = v;
      Pbf[(size_t)row * DDIM + col] = (bf16)v;
    }
  }
}

// ---------------------------------------------------------------------------
// Stats: per-row (max, sumexp) of logits Q @ K^T over all S columns.
// Grid: B * S/128 blocks; 8 waves each own 16 rows; stream K in 16-row tiles.
// Q fragments register-cached (invariant across the whole stream).
// ---------------------------------------------------------------------------
__global__ __launch_bounds__(256) void stats_kernel(
    const bf16* __restrict__ Qbf, const bf16* __restrict__ Kbf,
    float* __restrict__ mOut, float* __restrict__ lOut) {
  __shared__ bf16 sQ[128 * QSTR];
  __shared__ bf16 sX[16 * QSTR];

  const int tid = threadIdx.x;
  const int wave = tid >> 5, lane = tid & 31;
  const int lm = lane & 15, kg = lane >> 4;
  const int b = blockIdx.x >> 4, rb = blockIdx.x & 15;
  const int m0 = rb * 128;
  const bf16* Qb = Qbf + (size_t)b * S_LEN * DDIM;
  const bf16* Kb = Kbf + (size_t)b * S_LEN * DDIM;

#pragma unroll
  for (int it = 0; it < 16; ++it) {
    int flat = it * 2048 + tid * 8;
    int r = flat >> 8, c = flat & 255;
    *(bf16x8*)&sQ[r * QSTR + c] = *(const bf16x8*)(Qb + (size_t)(m0 + r) * DDIM + c);
  }
  __syncthreads();
  bf16x16 qfr[8];
#pragma unroll
  for (int kk = 0; kk < 8; ++kk) qfr[kk] = load_frag(sQ, 16 * wave, kk * 32, QSTR);

  float mloc[8], lloc[8];
#pragma unroll
  for (int r = 0; r < 8; ++r) { mloc[r] = -3.0e38f; lloc[r] = 0.f; }

  for (int j0 = 0; j0 < S_LEN; j0 += 16) {
    __syncthreads();
#pragma unroll
    for (int it = 0; it < 2; ++it) {
      int flat = it * 2048 + tid * 8;
      int r = flat >> 8, c = flat & 255;
      *(bf16x8*)&sX[r * QSTR + c] = *(const bf16x8*)(Kb + (size_t)(j0 + r) * DDIM + c);
    }
    if (j0 + 16 < S_LEN)
      __builtin_prefetch(Kb + (size_t)(j0 + 16) * DDIM + tid * 16, 0, 1);
    __syncthreads();
    f32x8 sacc = {};
#pragma unroll
    for (int kk = 0; kk < 8; ++kk) {
      bf16x16 bfr = load_frag(sX, 0, kk * 32, QSTR);
      sacc = wmma_bf16(qfr[kk], bfr, sacc);
    }
#pragma unroll
    for (int r = 0; r < 8; ++r) {
      float s = sacc[r];
      float mn = fmaxf(mloc[r], s);
      lloc[r] = lloc[r] * __expf(mloc[r] - mn) + __expf(s - mn);
      mloc[r] = mn;
    }
  }
  // merge across the 16 lanes holding the same M (C layout half-groups)
#pragma unroll
  for (int off = 1; off < 16; off <<= 1) {
#pragma unroll
    for (int r = 0; r < 8; ++r) {
      float mo = __shfl_xor(mloc[r], off, 32);
      float lo = __shfl_xor(lloc[r], off, 32);
      float mn = fmaxf(mloc[r], mo);
      lloc[r] = lloc[r] * __expf(mloc[r] - mn) + lo * __expf(mo - mn);
      mloc[r] = mn;
    }
  }
  if (lm == 0) {
    size_t base = (size_t)b * S_LEN + m0 + 16 * wave + 8 * kg;
#pragma unroll
    for (int r = 0; r < 8; ++r) { mOut[base + r] = mloc[r]; lOut[base + r] = lloc[r]; }
  }
}

// ---------------------------------------------------------------------------
// Fused: Out rows = Resid + a1 * softmaxed(Q.X1^T) @ X1 + a2 * (...X2...) @ X2
// byRow=1 -> forward product (stats indexed by output row)
// byRow=0 -> transposed product (stats indexed by streamed column)
// Grid: B * S/64 blocks; 8 waves = 4 row-blocks x 2 halves (j for GEMM1,
// d for GEMM2). Q fragments register-cached across both products.
// ---------------------------------------------------------------------------
__global__ __launch_bounds__(256) void fused_kernel(
    const bf16* __restrict__ Qbf, const float* __restrict__ Resid,
    const bf16* __restrict__ X1g, const float* __restrict__ m1,
    const float* __restrict__ l1, const float* __restrict__ a1p, int byRow1,
    const bf16* __restrict__ X2g, const float* __restrict__ m2,
    const float* __restrict__ l2, const float* __restrict__ a2p, int byRow2,
    float* __restrict__ Out, int outOff) {
  __shared__ bf16 sQ[64 * QSTR];      // Q rows (reused across both products)
  __shared__ bf16 sX[32 * QSTR];      // streamed tile, row-major (GEMM1 B^T)
  __shared__ bf16 sXT[DDIM * TSTR];   // streamed tile, transposed (GEMM2 B^T)
  __shared__ bf16 sP[64 * TSTR];      // normalized weights (GEMM2 A)

  const int tid = threadIdx.x;
  const int wave = tid >> 5, lane = tid & 31;
  const int lm = lane & 15, kg = lane >> 4;
  const int wm = wave >> 1, wn = wave & 1;
  const int b = blockIdx.x >> 5, rb = blockIdx.x & 31;
  const int m0 = rb * 64;
  const size_t bsOff = (size_t)b * S_LEN * DDIM;
  const size_t bRow = (size_t)b * S_LEN;

  const bf16* Qb = Qbf + bsOff;
#pragma unroll
  for (int it = 0; it < 8; ++it) {
    int flat = it * 2048 + tid * 8;
    int r = flat >> 8, c = flat & 255;
    *(bf16x8*)&sQ[r * QSTR + c] = *(const bf16x8*)(Qb + (size_t)(m0 + r) * DDIM + c);
  }
  __syncthreads();
  // Q fragments are invariant across all stream steps and both products.
  bf16x16 qfr[8];
#pragma unroll
  for (int kk = 0; kk < 8; ++kk) qfr[kk] = load_frag(sQ, 16 * wm, kk * 32, QSTR);

  f32x8 acc[8] = {};

  const bf16* Xs[2] = {X1g + bsOff, X2g + bsOff};
  const float* ms[2] = {m1 + bRow, m2 + bRow};
  const float* ls[2] = {l1 + bRow, l2 + bRow};
  const float al[2] = {a1p[0], a2p[0]};
  const int br[2] = {byRow1, byRow2};

  for (int p = 0; p < 2; ++p) {
    const bf16* X = Xs[p];
    const float* mst = ms[p];
    const float* lst = ls[p];
    const float alpha = al[p];
    const int byRow = br[p];

    float mrow[8], lrow[8];
    if (byRow) {
#pragma unroll
      for (int r = 0; r < 8; ++r) {
        int row = m0 + 16 * wm + r + 8 * kg;
        mrow[r] = mst[row];
        lrow[r] = alpha / lst[row];
      }
    }

    for (int j0 = 0; j0 < S_LEN; j0 += 32) {
      __syncthreads();  // prior step done with sX/sXT/sP
      // stream tile: 32 rows of X, row-major
#pragma unroll
      for (int it = 0; it < 4; ++it) {
        int flat = it * 2048 + tid * 8;
        int r = flat >> 8, c = flat & 255;
        *(bf16x8*)&sX[r * QSTR + c] =
            *(const bf16x8*)(X + (size_t)(j0 + r) * DDIM + c);
      }
      // prefetch next stream tile (one 128B line per lane covers the tile)
      if (j0 + 32 < S_LEN)
        __builtin_prefetch(X + (size_t)(j0 + 32) * DDIM + tid * 64, 0, 1);
      __syncthreads();
      transpose_tile(sX, sXT);
      // GEMM1: 16x16 logit tile per wave (K = 256), A cached in registers
      f32x8 sacc = {};
#pragma unroll
      for (int kk = 0; kk < 8; ++kk) {
        bf16x16 bfr = load_frag(sX, 16 * wn, kk * 32, QSTR);
        sacc = wmma_bf16(qfr[kk], bfr, sacc);
      }
      // finalize softmax weights (fold alpha) and stash bf16 P tile
      float mj = 0.f, lj = 0.f;
      if (!byRow) {
        int j = j0 + 16 * wn + lm;
        mj = mst[j];
        lj = alpha / lst[j];
      }
#pragma unroll
      for (int r = 0; r < 8; ++r) {
        float w = byRow ? __expf(sacc[r] - mrow[r]) * lrow[r]
                        : __expf(sacc[r] - mj) * lj;
        sP[(16 * wm + r + 8 * kg) * TSTR + (16 * wn + lm)] = (bf16)w;
      }
      __syncthreads();
      // GEMM2: acc[16 x 128] += P[16 x 32] @ X[32 x 128-half]
      bf16x16 pfr = load_frag(sP, 16 * wm, 0, TSTR);
#pragma unroll
      for (int t = 0; t < 8; ++t) {
        bf16x16 xf = load_frag(sXT, 128 * wn + 16 * t, 0, TSTR);
        acc[t] = wmma_bf16(pfr, xf, acc[t]);
      }
    }
  }

  // writeback: residual + accumulated products into the [B,S,768] slice
  float* Ob = Out + (size_t)b * S_LEN * (3 * DDIM) + outOff;
  const float* Rb = Resid + bsOff;
#pragma unroll
  for (int t = 0; t < 8; ++t) {
#pragma unroll
    for (int r = 0; r < 8; ++r) {
      int row = m0 + 16 * wm + r + 8 * kg;
      int col = 128 * wn + 16 * t + lm;
      Ob[(size_t)row * (3 * DDIM) + col] =
          Rb[(size_t)row * DDIM + col] + acc[t][r];
    }
  }
}

// ---------------------------------------------------------------------------
extern "C" void kernel_launch(void* const* d_in, const int* in_sizes, int n_in,
                              void* d_out, int out_size, void* d_ws,
                              size_t ws_size, hipStream_t stream) {
  (void)in_sizes; (void)n_in; (void)out_size; (void)ws_size;
  const int B = 8, DV = 1024, DT = 768, DA = 512;
  const size_t n = (size_t)B * S_LEN * DDIM;  // 4,194,304 elems per modality

  const float* video = (const float*)d_in[0];
  const float* text  = (const float*)d_in[1];
  const float* audio = (const float*)d_in[2];
  const float* Wv = (const float*)d_in[3];
  const float* bv = (const float*)d_in[4];
  const float* Wt = (const float*)d_in[5];
  const float* bt = (const float*)d_in[6];
  const float* Wa = (const float*)d_in[7];
  const float* ba = (const float*)d_in[8];
  const float* avt = (const float*)d_in[9];
  const float* ava = (const float*)d_in[10];
  const float* ata = (const float*)d_in[11];
  float* out = (float*)d_out;

  // workspace carving (~76 MB)
  char* ws = (char*)d_ws;
  float* v32 = (float*)ws; ws += n * 4;
  float* t32 = (float*)ws; ws += n * 4;
  float* a32 = (float*)ws; ws += n * 4;
  bf16* vbf = (bf16*)ws; ws += n * 2;
  bf16* tbf = (bf16*)ws; ws += n * 2;
  bf16* abf = (bf16*)ws; ws += n * 2;
  const size_t rs = (size_t)B * S_LEN * 4;
  float* mvt = (float*)ws; ws += rs;  float* lvt = (float*)ws; ws += rs;
  float* mva = (float*)ws; ws += rs;  float* lva = (float*)ws; ws += rs;
  float* mta = (float*)ws; ws += rs;  float* lta = (float*)ws; ws += rs;

  dim3 blk(256);
  // projections
  proj_kernel<<<256, blk, 0, stream>>>(video, Wv, bv, DV, v32, vbf);
  proj_kernel<<<256, blk, 0, stream>>>(text,  Wt, bt, DT, t32, tbf);
  proj_kernel<<<256, blk, 0, stream>>>(audio, Wa, ba, DA, a32, abf);
  // softmax row stats for the three logit matrices
  stats_kernel<<<128, blk, 0, stream>>>(vbf, tbf, mvt, lvt);
  stats_kernel<<<128, blk, 0, stream>>>(vbf, abf, mva, lva);
  stats_kernel<<<128, blk, 0, stream>>>(tbf, abf, mta, lta);
  // fused outputs (forward products byRow=1, transposed byRow=0)
  fused_kernel<<<256, blk, 0, stream>>>(vbf, v32,
      tbf, mvt, lvt, avt, 1,  abf, mva, lva, ava, 1,  out, 0);
  fused_kernel<<<256, blk, 0, stream>>>(tbf, t32,
      vbf, mvt, lvt, avt, 0,  abf, mta, lta, ata, 1,  out, DDIM);
  fused_kernel<<<256, blk, 0, stream>>>(abf, a32,
      vbf, mva, lva, ava, 0,  tbf, mta, lta, ata, 0,  out, 2 * DDIM);
}
